// LocalWindowAttentionBlock_49460843380890
// MI455X (gfx1250) — compile-verified, matrix operand order checked by reading, not run on hardware
//
#include <hip/hip_runtime.h>
#include <hip/hip_bf16.h>
#include <math.h>

// ---------------- problem constants ----------------
#define L_SEQ  4096
#define DMODEL 256
#define NHEAD  8
#define DHEAD  32
#define FFDIM  1024
#define WINSZ  128
// padded row length for V^T: 128 left pad + 4096 + 160 right pad
#define LDV    4384
#define VPAD   128
// K-chunk staged into LDS per GEMM iteration
#define KCHUNK 128

typedef __attribute__((ext_vector_type(16))) _Float16 v16h;
typedef __attribute__((ext_vector_type(8)))  float    v8f;
typedef unsigned int v4u __attribute__((ext_vector_type(4)));
typedef int          v4i __attribute__((ext_vector_type(4)));
typedef int          v8i __attribute__((ext_vector_type(8)));

#if defined(__has_builtin)
#  if __has_builtin(__builtin_amdgcn_tensor_load_to_lds)
#    define HAVE_TDM 1
#  else
#    define HAVE_TDM 0
#  endif
#else
#  define HAVE_TDM 0
#endif

// -----------------------------------------------------------------------------
// Fragment loader for V_WMMA_F32_16X16X32_F16 operands.
// Source is a row-major [16 rows][K-slice of 32] tile with leading dim `ld`.
// Works for both A (rows = M) and B (rows = N, i.e. weight stored [N][K]).
// ISA layout (16-bit A 16x32): lanes 0-15 -> K base 0, lanes 16-31 -> K base 8;
// vgpr pair p<4: K = kb+2p, p>=4: K = kb+16+2(p-4)  => two contiguous 16B runs.
// -----------------------------------------------------------------------------
__device__ __forceinline__ v16h load_frag16(const _Float16* base, int ld) {
  const int lane = threadIdx.x & 31;
  const int r    = lane & 15;
  const int kb   = (lane & 16) ? 8 : 0;
  const _Float16* row = base + r * ld;
  v16h f;
#pragma unroll
  for (int p = 0; p < 8; ++p) {
    const int k = (p < 4) ? (kb + 2 * p) : (kb + 16 + 2 * (p - 4));
    f[2 * p]     = row[k];
    f[2 * p + 1] = row[k + 1];
  }
  return f;
}

#if HAVE_TDM
// -----------------------------------------------------------------------------
// Issue one TDM 2D tile load: rows x cols f16 elements, global row stride
// ldg elements, packed contiguously into LDS at `lds_dst`.
// Descriptor built per CDNA5 ISA section 8 (D# groups 0..3).
// -----------------------------------------------------------------------------
__device__ __forceinline__ void tdm_load_2d_f16(const _Float16* gsrc,
                                                const _Float16* lds_dst,
                                                unsigned rows, unsigned cols,
                                                unsigned ldg) {
  const unsigned long long ga = (unsigned long long)(uintptr_t)gsrc;
  const unsigned lds_off = (unsigned)(uintptr_t)lds_dst;  // LDS addr = addr[31:0]

  v4u g0;
  g0[0] = 1u;                                            // count=1, user D#
  g0[1] = lds_off;                                       // lds_addr
  g0[2] = (unsigned)(ga & 0xFFFFFFFFull);                // global_addr[31:0]
  g0[3] = (unsigned)((ga >> 32) & 0x1FFFFFFull)          // global_addr[56:32]
        | (2u << 30);                                    // type = 2 ("image")

  v8i g1;
  g1[0] = (int)(1u << 16);                               // data_size=1 (2B), mask=0
  g1[1] = (int)((cols & 0xFFFFu) << 16);                 // tensor_dim0[15:0]
  g1[2] = (int)((cols >> 16) | ((rows & 0xFFFFu) << 16));// td0[31:16] | td1[15:0]
  g1[3] = (int)((rows >> 16) | (cols << 16));            // td1[31:16] | tile_dim0
  g1[4] = (int)rows;                                     // tile_dim1 (tile_dim2=0)
  g1[5] = (int)ldg;                                      // tensor_dim0_stride[31:0]
  g1[6] = 0;                                             // stride[47:32] | td1_stride lo
  g1[7] = 0;

  v4i g2 = {0, 0, 0, 0};                                 // 2D: dims 2/3 unused
  v4i g3 = {0, 0, 0, 0};

#if __clang_major__ >= 23
  v8i g4 = {0, 0, 0, 0, 0, 0, 0, 0};
  __builtin_amdgcn_tensor_load_to_lds(g0, g1, g2, g3, g4, 0);
#else
  __builtin_amdgcn_tensor_load_to_lds(g0, g1, g2, g3, 0);
#endif
}
#endif  // HAVE_TDM

// ---------------- small utility kernels ----------------
__global__ void cvt_f32_to_f16_kernel(const float* __restrict__ in,
                                      _Float16* __restrict__ out, int n) {
  int i = blockIdx.x * 256 + threadIdx.x;
  if (i < n) out[i] = (_Float16)in[i];
}

__global__ void zero_f16_kernel(_Float16* __restrict__ p, int n) {
  int i = blockIdx.x * 256 + threadIdx.x;
  if (i < n) p[i] = (_Float16)0.0f;
}

// LayerNorm over D=256, one row per block (256 threads), output f16.
__global__ void ln_f16_kernel(const float* __restrict__ x,
                              const float* __restrict__ g,
                              const float* __restrict__ b,
                              _Float16* __restrict__ out) {
  __shared__ float red[DMODEL];
  const int row = blockIdx.x;
  const int t   = threadIdx.x;
  const float v = x[row * DMODEL + t];
  red[t] = v; __syncthreads();
  for (int s = DMODEL / 2; s > 0; s >>= 1) {
    if (t < s) red[t] += red[t + s];
    __syncthreads();
  }
  const float mu = red[0] * (1.0f / DMODEL);
  __syncthreads();
  const float d = v - mu;
  red[t] = d * d; __syncthreads();
  for (int s = DMODEL / 2; s > 0; s >>= 1) {
    if (t < s) red[t] += red[t + s];
    __syncthreads();
  }
  const float var = red[0] * (1.0f / DMODEL);
  const float y = d * rsqrtf(var + 1e-5f) * g[t] + b[t];
  out[row * DMODEL + t] = (_Float16)y;
}

// -----------------------------------------------------------------------------
// WMMA GEMM with TDM-staged LDS tiles:
//   C[M=4096][N] = A[M][K](f16) x Bw[N][K]^T(f16) + bias
// block = 128 threads (4 waves); block tile = 16 (M) x 64 (N); wave w owns the
// 16x16 tile at n0 = 64*blockIdx.y + 16*w. Per K-chunk of 128:
//   wave0 issues 2 tensor_load_to_lds (A tile 16x128, B tile 64x128),
//   s_wait_tensorcnt 0, workgroup barrier, then 4 WMMAs from LDS fragments.
// mode 0: QKV epilogue (scale+scatter Q, scatter K, scatter V^T padded)
// mode 1: f32 out = acc + bias + res        (out-proj, FFN2 residual adds)
// mode 2: f16 out = GELU_exact(acc + bias)  (FFN1)
// -----------------------------------------------------------------------------
__global__ void gemm_wmma_kernel(const _Float16* __restrict__ A,
                                 const _Float16* __restrict__ Bw,
                                 const float* __restrict__ bias,
                                 int N, int K, int mode,
                                 _Float16* __restrict__ out_q,
                                 _Float16* __restrict__ out_k,
                                 _Float16* __restrict__ out_vT,
                                 float* __restrict__ out_f32,
                                 const float* __restrict__ res,
                                 _Float16* __restrict__ out_f16) {
  __shared__ _Float16 sA[16 * KCHUNK];   //  4 KB
  __shared__ _Float16 sB[64 * KCHUNK];   // 16 KB

  const int wave = threadIdx.x >> 5;
  const int lane = threadIdx.x & 31;
  const int m0 = blockIdx.x * 16;
  const int n0 = blockIdx.y * 64 + wave * 16;

  const _Float16* Ablk = A + (size_t)m0 * K;                    // 16 rows
  const _Float16* Bblk = Bw + (size_t)(blockIdx.y * 64) * K;    // 64 rows

  v8f acc = {};
  for (int kc = 0; kc < K; kc += KCHUNK) {
    __syncthreads();   // previous chunk fully consumed before overwrite
#if HAVE_TDM
    if (threadIdx.x < 32) {
      tdm_load_2d_f16(Ablk + kc, sA, 16, KCHUNK, (unsigned)K);
      tdm_load_2d_f16(Bblk + kc, sB, 64, KCHUNK, (unsigned)K);
      __builtin_amdgcn_s_wait_tensorcnt(0);
    }
#else
    // cooperative fallback: 16B vector copies, 80 rows x 128 cols
    {
      const int R4 = KCHUNK / 8;  // uint4 per row
      for (int i = threadIdx.x; i < 16 * R4; i += 128) {
        const int r = i / R4, c = i % R4;
        ((uint4*)sA)[i] = ((const uint4*)(Ablk + (size_t)r * K + kc))[c];
      }
      for (int i = threadIdx.x; i < 64 * R4; i += 128) {
        const int r = i / R4, c = i % R4;
        ((uint4*)sB)[i] = ((const uint4*)(Bblk + (size_t)r * K + kc))[c];
      }
    }
#endif
    if (kc + KCHUNK < K)
      __builtin_prefetch(Bblk + (size_t)kc + KCHUNK, 0, 1);  // global_prefetch_b8
    __syncthreads();

#pragma unroll
    for (int k = 0; k < KCHUNK; k += 32) {
      v16h a = load_frag16(sA + k, KCHUNK);
      v16h b = load_frag16(sB + wave * 16 * KCHUNK + k, KCHUNK);
      acc = __builtin_amdgcn_wmma_f32_16x16x32_f16(false, a, false, b,
                                                   (short)0, acc, false, false);
    }
  }

  const int n   = n0 + (lane & 15);
  const int mro = (lane & 16) ? 8 : 0;
  const float bv = bias[n];

  if (mode == 0) {
#pragma unroll
    for (int r = 0; r < 8; ++r) {
      const int row = m0 + mro + r;
      const float v = acc[r] + bv;
      if (n < DMODEL) {
        out_q[(size_t)row * DMODEL + n] = (_Float16)(v * 0.17677669529663687f);
      } else if (n < 2 * DMODEL) {
        out_k[(size_t)row * DMODEL + (n - DMODEL)] = (_Float16)v;
      } else {
        out_vT[(size_t)(n - 2 * DMODEL) * LDV + VPAD + row] = (_Float16)v;
      }
    }
  } else if (mode == 1) {
#pragma unroll
    for (int r = 0; r < 8; ++r) {
      const int row = m0 + mro + r;
      out_f32[(size_t)row * N + n] = acc[r] + bv + res[(size_t)row * N + n];
    }
  } else {
#pragma unroll
    for (int r = 0; r < 8; ++r) {
      const int row = m0 + mro + r;
      const float v = acc[r] + bv;
      const float g = 0.5f * v * (1.0f + erff(v * 0.7071067811865475f));
      out_f16[(size_t)row * N + n] = (_Float16)g;
    }
  }
}

// -----------------------------------------------------------------------------
// Local-window attention. One wave (32 threads) per (16-query tile, head).
// Window |i-j| <= 128 -> union over a 16-query tile spans 272 keys = 17 tiles.
// scores: 17 WMMAs (K = DH = 32 exactly). Softmax staged in LDS (f32), P in
// f16 (A-fragment layout via LDS reload), PV: 9 chunks x 2 WMMAs. Q is
// pre-scaled; V^T is zero-padded so OOB chunks multiply zero probabilities.
// -----------------------------------------------------------------------------
__global__ void attn_wmma_kernel(const _Float16* __restrict__ q,
                                 const _Float16* __restrict__ k,
                                 const _Float16* __restrict__ vT,
                                 _Float16* __restrict__ o) {
  __shared__ float    sS[16 * 272];
  __shared__ _Float16 sP[16 * 288];   // 272 padded to 288 (9 chunks of 32)
  __shared__ float    sInv[16];

  const int q0   = blockIdx.x * 16;
  const int h    = blockIdx.y;
  const int lane = threadIdx.x;
  const int t0   = q0 - WINSZ;                 // first (possibly OOB) key tile
  const int n    = lane & 15;
  const int mro  = (lane & 16) ? 8 : 0;

  const v16h aq = load_frag16(q + (size_t)q0 * DMODEL + h * DHEAD, DMODEL);

  for (int t = 0; t < 17; ++t) {
    const int kt = t0 + t * 16;                // uniform (blockIdx-derived)
    const bool tile_ok = (kt >= 0) && (kt <= L_SEQ - 16);
    v8f c = {};
    if (tile_ok) {
      v16h bk = load_frag16(k + (size_t)kt * DMODEL + h * DHEAD, DMODEL);
      c = __builtin_amdgcn_wmma_f32_16x16x32_f16(false, aq, false, bk,
                                                 (short)0, c, false, false);
    }
#pragma unroll
    for (int r = 0; r < 8; ++r) {
      const int m  = mro + r;
      const int qi = q0 + m;
      const int ki = kt + n;
      int d = qi - ki; if (d < 0) d = -d;
      const bool ok = tile_ok && (d <= WINSZ);
      sS[m * 272 + t * 16 + n] = ok ? c[r] : -1e30f;
    }
  }
  __syncthreads();

  if (lane < 16) {
    float mx = -1e30f;
    for (int j = 0; j < 272; ++j) mx = fmaxf(mx, sS[lane * 272 + j]);
    float sum = 0.0f;
    for (int j = 0; j < 272; ++j) {
      const float e = __expf(sS[lane * 272 + j] - mx);
      sum += e;
      sP[lane * 288 + j] = (_Float16)e;
    }
    for (int j = 272; j < 288; ++j) sP[lane * 288 + j] = (_Float16)0.0f;
    sInv[lane] = 1.0f / sum;
  }
  __syncthreads();

  v8f o0 = {}, o1 = {};
  for (int c0 = 0; c0 < 288; c0 += 32) {
    v16h ap = load_frag16(&sP[0] + c0, 288);
    const _Float16* vb = vT + (size_t)(h * DHEAD) * LDV + VPAD + t0 + c0;
    v16h b0 = load_frag16(vb, LDV);             // dh 0..15
    v16h b1 = load_frag16(vb + 16 * LDV, LDV);  // dh 16..31
    o0 = __builtin_amdgcn_wmma_f32_16x16x32_f16(false, ap, false, b0,
                                                (short)0, o0, false, false);
    o1 = __builtin_amdgcn_wmma_f32_16x16x32_f16(false, ap, false, b1,
                                                (short)0, o1, false, false);
  }

#pragma unroll
  for (int r = 0; r < 8; ++r) {
    const int m   = mro + r;
    const float s = sInv[m];
    const int row = q0 + m;
    o[(size_t)row * DMODEL + h * DHEAD + n]      = (_Float16)(o0[r] * s);
    o[(size_t)row * DMODEL + h * DHEAD + 16 + n] = (_Float16)(o1[r] * s);
  }
}

// -----------------------------------------------------------------------------
extern "C" void kernel_launch(void* const* d_in, const int* in_sizes, int n_in,
                              void* d_out, int out_size, void* d_ws, size_t ws_size,
                              hipStream_t stream) {
  (void)in_sizes; (void)n_in; (void)out_size; (void)ws_size;

  const float* x         = (const float*)d_in[0];
  const float* ln1_g     = (const float*)d_in[1];
  const float* ln1_b     = (const float*)d_in[2];
  const float* ln2_g     = (const float*)d_in[3];
  const float* ln2_b     = (const float*)d_in[4];
  const float* in_proj_w = (const float*)d_in[5];
  const float* in_proj_b = (const float*)d_in[6];
  const float* out_w     = (const float*)d_in[7];
  const float* out_b     = (const float*)d_in[8];
  const float* w1        = (const float*)d_in[9];
  const float* b1        = (const float*)d_in[10];
  const float* w2        = (const float*)d_in[11];
  const float* b2        = (const float*)d_in[12];
  float* out = (float*)d_out;

  // ---- scratch carve (~26 MB) ----
  char* p = (char*)d_ws;
  auto take = [&](size_t bytes) {
    char* r = p;
    p += (bytes + 255) & ~(size_t)255;
    return r;
  };
  _Float16* h1   = (_Float16*)take((size_t)L_SEQ * DMODEL * 2);
  _Float16* h2   = (_Float16*)take((size_t)L_SEQ * DMODEL * 2);
  _Float16* q16  = (_Float16*)take((size_t)L_SEQ * DMODEL * 2);
  _Float16* k16  = (_Float16*)take((size_t)L_SEQ * DMODEL * 2);
  _Float16* vT   = (_Float16*)take((size_t)DMODEL * LDV * 2);
  _Float16* o16  = (_Float16*)take((size_t)L_SEQ * DMODEL * 2);
  _Float16* a1   = (_Float16*)take((size_t)L_SEQ * FFDIM * 2);
  float*    x1   = (float*)take((size_t)L_SEQ * DMODEL * 4);
  _Float16* wqkv = (_Float16*)take((size_t)3 * DMODEL * DMODEL * 2);
  _Float16* wout = (_Float16*)take((size_t)DMODEL * DMODEL * 2);
  _Float16* w1h  = (_Float16*)take((size_t)FFDIM * DMODEL * 2);
  _Float16* w2h  = (_Float16*)take((size_t)DMODEL * FFDIM * 2);

  // ---- 1) weights -> f16 (already [N][K] row-major) ----
  int nw;
  nw = 3 * DMODEL * DMODEL;
  cvt_f32_to_f16_kernel<<<(nw + 255) / 256, 256, 0, stream>>>(in_proj_w, wqkv, nw);
  nw = DMODEL * DMODEL;
  cvt_f32_to_f16_kernel<<<(nw + 255) / 256, 256, 0, stream>>>(out_w, wout, nw);
  nw = FFDIM * DMODEL;
  cvt_f32_to_f16_kernel<<<(nw + 255) / 256, 256, 0, stream>>>(w1, w1h, nw);
  nw = DMODEL * FFDIM;
  cvt_f32_to_f16_kernel<<<(nw + 255) / 256, 256, 0, stream>>>(w2, w2h, nw);

  // zero padded V^T (pads must be exact zeros under zero probabilities)
  nw = DMODEL * LDV;
  zero_f16_kernel<<<(nw + 255) / 256, 256, 0, stream>>>(vT, nw);

  // ---- 2) LN1 -> f16 ----
  ln_f16_kernel<<<L_SEQ, DMODEL, 0, stream>>>(x, ln1_g, ln1_b, h1);

  // ---- 3) QKV projection (N=768, K=256), scatter Q/K/V^T ----
  gemm_wmma_kernel<<<dim3(L_SEQ / 16, (3 * DMODEL) / 64), 128, 0, stream>>>(
      h1, wqkv, in_proj_b, 3 * DMODEL, DMODEL, 0,
      q16, k16, vT, nullptr, nullptr, nullptr);

  // ---- 4) local-window attention ----
  attn_wmma_kernel<<<dim3(L_SEQ / 16, NHEAD), 32, 0, stream>>>(q16, k16, vT, o16);

  // ---- 5) out-proj + residual -> x1 (f32) ----
  gemm_wmma_kernel<<<dim3(L_SEQ / 16, DMODEL / 64), 128, 0, stream>>>(
      o16, wout, out_b, DMODEL, DMODEL, 1,
      nullptr, nullptr, nullptr, x1, x, nullptr);

  // ---- 6) LN2 -> f16 ----
  ln_f16_kernel<<<L_SEQ, DMODEL, 0, stream>>>(x1, ln2_g, ln2_b, h2);

  // ---- 7) FFN1 + exact GELU -> a1 (f16) ----
  gemm_wmma_kernel<<<dim3(L_SEQ / 16, FFDIM / 64), 128, 0, stream>>>(
      h2, w1h, b1, FFDIM, DMODEL, 2,
      nullptr, nullptr, nullptr, nullptr, nullptr, a1);

  // ---- 8) FFN2 + residual -> d_out (f32) ----
  gemm_wmma_kernel<<<dim3(L_SEQ / 16, DMODEL / 64), 128, 0, stream>>>(
      a1, w2h, b2, DMODEL, FFDIM, 1,
      nullptr, nullptr, nullptr, out, x1, nullptr);
}
